// pocket_att_net2_82832739270708
// MI455X (gfx1250) — matrix operand order repaired and checked by reading, not compiled, not run on hardware
//
#include <hip/hip_runtime.h>
#include <stdint.h>

// ============================================================================
// pocket_att_net2 for MI455X (gfx1250).
//  - All GEMMs: v_wmma_f32_16x16x32_bf16 (wave32, f32 accum), 4-way N-blocking
//    per wave so the A operand (incl. the 134 MB attention matrix, L2-resident)
//    is loaded once per 4 WMMAs.
//  - Energy kernels stage the p1 B-panel in LDS via the Tensor Data Mover
//    (tensor_load_to_lds + s_wait_tensorcnt, double buffered), shared by all
//    8 waves of the workgroup; falls back to cooperative copies if the TDM
//    builtin is unavailable.
//  - Softmax: pass 1 online (max,sumexp) + shfl reduction; pass 2 recomputes
//    energy and writes att row-major + transposed (packed 16B lane stores).
// ============================================================================

typedef unsigned short u16;                       // raw bf16 storage
typedef __attribute__((ext_vector_type(16))) __bf16 v16bf;
typedef __attribute__((ext_vector_type(8)))  float  v8f;
typedef __attribute__((ext_vector_type(4)))  unsigned int v4u;
typedef __attribute__((ext_vector_type(8)))  int v8i;
typedef __attribute__((ext_vector_type(4)))  int v4i;

#define B_    16
#define NL    256
#define NP    2048
#define HH    8
#define TPB   256           // 8 waves per block
#define CHUNK 128           // p-rows staged in LDS per TDM transfer
#define NCH   (NP / CHUNK)  // 16 chunks

#if defined(__has_builtin)
#  if __has_builtin(__builtin_amdgcn_tensor_load_to_lds) && \
      __has_builtin(__builtin_amdgcn_s_wait_tensorcnt)
#    define HAVE_TDM 1
#  endif
#endif
#ifndef HAVE_TDM
#  define HAVE_TDM 0
#endif

__device__ __forceinline__ u16 f2bf(float f) {    // RNE f32->bf16
  uint32_t u = __builtin_bit_cast(uint32_t, f);
  u += 0x7FFFu + ((u >> 16) & 1u);
  return (u16)(u >> 16);
}

union V16U { struct { uint4 lo, hi; } q; v16bf v; };

// A tile (16x32 bf16) from row-major storage, lda elements.
__device__ __forceinline__ v16bf load_a16(const u16* base, int lda, int lr, int hf) {
  const u16* p = base + (size_t)lr * lda + hf * 8;
  V16U u;
  u.q.lo = *(const uint4*)(p);
  u.q.hi = *(const uint4*)(p + 16);
  return u.v;
}

// B tile (32x16 bf16) from N-major storage (row n = contiguous K run).
__device__ __forceinline__ v16bf load_b16(const u16* base, int ldb, int lr, int hf) {
  const u16* p = base + (size_t)lr * ldb + hf * 8;
  V16U u;
  u.q.lo = *(const uint4*)(p);
  u.q.hi = *(const uint4*)(p + 16);
  return u.v;
}

__device__ __forceinline__ v8f wmma_bf16(v16bf a, v16bf b, v8f c) {
  return __builtin_amdgcn_wmma_f32_16x16x32_bf16(false, a, false, b, (short)0, c,
                                                 false, false);
}

__device__ __forceinline__ float band_mask(float d, int h) {
  bool c = (h < 7) ? (d > (float)h && d <= (float)(h + 3)) : (d > 7.0f);
  return c ? (1.0f / d) : 0.0f;
}

#if HAVE_TDM
// TDM 1-D copy: CHUNK*64 bf16 (16 KB) global -> LDS. D# per ISA ch.8:
// g0 = {count=1, lds_addr, ga[31:0], ga[56:32]|type=2}
// g1 = {data_size=2B, tensor_dim0=8192, tensor_dim1=1, tile_dim0=8192 (1-D), stride}
// This toolchain's builtin takes 6 args (g0, g1, g2, g3, g4, cpol).
__device__ __forceinline__ void tdm_load_chunk(const u16* gsrc, void* lds_dst) {
  uint64_t ga = (uint64_t)(uintptr_t)gsrc;
  uint32_t la = (uint32_t)(uintptr_t)lds_dst;
  const uint32_t N = CHUNK * 64;  // 8192 elements
  v4u g0 = { 1u, la, (uint32_t)ga, (uint32_t)(ga >> 32) | (2u << 30) };
  v8i g1 = { (int)(1u << 16),                 // data_size=1 -> 2 bytes, no flags
             (int)((N & 0xFFFFu) << 16),      // tensor_dim0[15:0]
             (int)((N >> 16) | (1u << 16)),   // tensor_dim0[31:16] | tensor_dim1=1
             (int)(N << 16),                  // tile_dim0 (tile_dim1=0 -> 1-D)
             0, (int)N, 0, 0 };               // tensor_dim0_stride
  v4i z4 = { 0, 0, 0, 0 };
  v8i z8 = { 0, 0, 0, 0, 0, 0, 0, 0 };
  __builtin_amdgcn_tensor_load_to_lds(g0, g1, z4, z4, z8, 0);
}
#endif

// ---------------------------------------------------------------------------
__global__ void cvt_cat_kernel(const float* __restrict__ src, u16* __restrict__ cat,
                               int n) {
  int i = blockIdx.x * blockDim.x + threadIdx.x;
  if (i < n) {
    int row = i >> 6, d = i & 63;
    cat[(size_t)row * 128 + 64 + d] = f2bf(src[i]);
  }
}

__global__ void wt_t_kernel(const float* __restrict__ w, u16* __restrict__ wt,
                            int IN, int OUT) {
  int i = blockIdx.x * blockDim.x + threadIdx.x;
  if (i < IN * OUT) {
    int o = i / IN, k = i - o * IN;
    wt[(size_t)o * IN + k] = f2bf(w[(size_t)k * OUT + o]);
  }
}

// ---------------------------------------------------------------------------
// Projection: relu(x @ W + b). One wave per (b, mtile, head): A loaded once,
// 8 WMMA (4 n-tiles x K=64).
// ---------------------------------------------------------------------------
__global__ __launch_bounds__(TPB)
void proj_kernel(const u16* __restrict__ xb, const u16* __restrict__ wt,
                 const float* __restrict__ bias, int Ntok,
                 u16* __restrict__ outA, u16* __restrict__ outT) {
  int wid  = (blockIdx.x * TPB + threadIdx.x) >> 5;
  int lane = threadIdx.x & 31, lr = lane & 15, hf = lane >> 4;
  int mtpb = Ntok >> 4;
  if (wid >= B_ * mtpb * HH) return;
  int h  = wid & 7;
  int mt = (wid >> 3) % mtpb;
  int b  = wid / (8 * mtpb);
  int m0 = mt << 4;

  const u16* abase = xb + (size_t)(b * Ntok + m0) * 128;
  v16bf a0 = load_a16(abase,      128, lr, hf);
  v16bf a1 = load_a16(abase + 32, 128, lr, hf);

#pragma unroll
  for (int dt = 0; dt < 4; ++dt) {
    int n0 = h * 64 + dt * 16;
    const u16* bbase = wt + (size_t)n0 * 64;
    v8f acc = {};
    acc = wmma_bf16(a0, load_b16(bbase,      64, lr, hf), acc);
    acc = wmma_bf16(a1, load_b16(bbase + 32, 64, lr, hf), acc);
    float bv = bias[n0 + lr];
    u16 res[8];
#pragma unroll
    for (int r = 0; r < 8; ++r) res[r] = f2bf(fmaxf(acc[r] + bv, 0.0f));
    int d0 = dt * 16;
    if (outA) {  // head-major [b][h][m][d]
      size_t base = ((size_t)(b * HH + h)) * Ntok * 64;
#pragma unroll
      for (int r = 0; r < 8; ++r)
        outA[base + (size_t)(m0 + r + hf * 8) * 64 + d0 + lr] = res[r];
    } else {     // transposed [b][h][d][m]: packed 16B store
      union { uint4 q; u16 s[8]; } pk;
#pragma unroll
      for (int r = 0; r < 8; ++r) pk.s[r] = res[r];
      size_t off = (((size_t)(b * HH + h) * 64) + d0 + lr) * Ntok + m0 + hf * 8;
      *(uint4*)(outT + off) = pk.q;
    }
  }
}

// ---------------------------------------------------------------------------
// Energy pass 1: online softmax stats. Block = 8 waves sharing (b,h); p1
// panel staged in LDS (TDM double-buffer or cooperative copy fallback).
// ---------------------------------------------------------------------------
__global__ __launch_bounds__(TPB)
void att_stats_kernel(const u16* __restrict__ l1h, const u16* __restrict__ p1h,
                      const float* __restrict__ dist, float* __restrict__ stats) {
  __shared__ __align__(16) u16 pbuf[2][CHUNK * 64];   // 2 x 16 KB
  int tid  = threadIdx.x;
  int lane = tid & 31, lr = lane & 15, hf = lane >> 4, w = tid >> 5;
  int bh  = blockIdx.x >> 1;          // (b*8+h), 0..127
  int ltg = blockIdx.x & 1;
  int b = bh >> 3, h = bh & 7;
  int lt = ltg * 8 + w;

  const u16* abase = l1h + (((size_t)bh * NL) + lt * 16) * 64;
  v16bf a0 = load_a16(abase,      64, lr, hf);
  v16bf a1 = load_a16(abase + 32, 64, lr, hf);
  const float* drow = dist + ((size_t)b * NL + lt * 16) * NP;
  const u16* psrc = p1h + (size_t)bh * NP * 64;

  float mrow[8], srow[8];
#pragma unroll
  for (int r = 0; r < 8; ++r) { mrow[r] = -3.0e38f; srow[r] = 0.0f; }

#if HAVE_TDM
  if (tid < 32) tdm_load_chunk(psrc, &pbuf[0][0]);
#endif
  for (int c = 0; c < NCH; ++c) {
#if HAVE_TDM
    if (tid < 32) {
      if (c + 1 < NCH) {
        tdm_load_chunk(psrc + (size_t)(c + 1) * CHUNK * 64, &pbuf[(c + 1) & 1][0]);
        __builtin_amdgcn_s_wait_tensorcnt(1);   // chunk c complete (in-order)
      } else {
        __builtin_amdgcn_s_wait_tensorcnt(0);
      }
    }
    __syncthreads();
#else
    __syncthreads();
    {  // cooperative copy: 8192 u16 = 1024 uint4, 4 per thread
      const uint4* s = (const uint4*)(psrc + (size_t)c * CHUNK * 64);
      uint4* d = (uint4*)&pbuf[c & 1][0];
#pragma unroll
      for (int i = 0; i < 4; ++i) d[tid + i * TPB] = s[tid + i * TPB];
    }
    __syncthreads();
#endif
    const u16* bufp = &pbuf[c & 1][0];
    for (int ptl = 0; ptl < CHUNK / 16; ++ptl) {
      int p0 = c * CHUNK + ptl * 16;
      const u16* bb = bufp + (size_t)ptl * 16 * 64;
      v8f acc = {};
      acc = wmma_bf16(a0, load_b16(bb,      64, lr, hf), acc);   // ds_load_b128
      acc = wmma_bf16(a1, load_b16(bb + 32, 64, lr, hf), acc);
#pragma unroll
      for (int r = 0; r < 8; ++r) {
        float d  = drow[(size_t)(r + hf * 8) * NP + p0 + lr];
        float v  = acc[r] * 0.125f * band_mask(d, h);
        float nm = fmaxf(mrow[r], v);
        srow[r]  = srow[r] * __expf(mrow[r] - nm) + __expf(v - nm);
        mrow[r]  = nm;
      }
    }
    __syncthreads();
  }
#pragma unroll
  for (int off = 1; off < 16; off <<= 1) {
#pragma unroll
    for (int r = 0; r < 8; ++r) {
      float om = __shfl_xor(mrow[r], off, 32);
      float os = __shfl_xor(srow[r], off, 32);
      float nm = fmaxf(mrow[r], om);
      srow[r]  = srow[r] * __expf(mrow[r] - nm) + os * __expf(om - nm);
      mrow[r]  = nm;
    }
  }
  if (lr == 0) {
#pragma unroll
    for (int r = 0; r < 8; ++r) {
      size_t row = ((size_t)bh * NL) + lt * 16 + r + hf * 8;
      stats[row * 2 + 0] = mrow[r];
      stats[row * 2 + 1] = srow[r];
    }
  }
}

// ---------------------------------------------------------------------------
// Energy pass 2: recompute, normalize, write att + att_t. Same TDM staging.
// ---------------------------------------------------------------------------
__global__ __launch_bounds__(TPB)
void att_write_kernel(const u16* __restrict__ l1h, const u16* __restrict__ p1h,
                      const float* __restrict__ dist, const float* __restrict__ stats,
                      u16* __restrict__ att, u16* __restrict__ att_t) {
  __shared__ __align__(16) u16 pbuf[2][CHUNK * 64];
  int tid  = threadIdx.x;
  int lane = tid & 31, lr = lane & 15, hf = lane >> 4, w = tid >> 5;
  int bh  = blockIdx.x >> 1;
  int ltg = blockIdx.x & 1;
  int b = bh >> 3, h = bh & 7;
  int lt = ltg * 8 + w;

  const u16* abase = l1h + (((size_t)bh * NL) + lt * 16) * 64;
  v16bf a0 = load_a16(abase,      64, lr, hf);
  v16bf a1 = load_a16(abase + 32, 64, lr, hf);
  const float* drow = dist + ((size_t)b * NL + lt * 16) * NP;
  const u16* psrc = p1h + (size_t)bh * NP * 64;

  float mrow[8], invs[8];
#pragma unroll
  for (int r = 0; r < 8; ++r) {
    size_t row = ((size_t)bh * NL) + lt * 16 + r + hf * 8;
    mrow[r] = stats[row * 2 + 0];
    invs[r] = 1.0f / stats[row * 2 + 1];
  }
  size_t arow = ((size_t)bh * NL) + lt * 16;

#if HAVE_TDM
  if (tid < 32) tdm_load_chunk(psrc, &pbuf[0][0]);
#endif
  for (int c = 0; c < NCH; ++c) {
#if HAVE_TDM
    if (tid < 32) {
      if (c + 1 < NCH) {
        tdm_load_chunk(psrc + (size_t)(c + 1) * CHUNK * 64, &pbuf[(c + 1) & 1][0]);
        __builtin_amdgcn_s_wait_tensorcnt(1);
      } else {
        __builtin_amdgcn_s_wait_tensorcnt(0);
      }
    }
    __syncthreads();
#else
    __syncthreads();
    {
      const uint4* s = (const uint4*)(psrc + (size_t)c * CHUNK * 64);
      uint4* d = (uint4*)&pbuf[c & 1][0];
#pragma unroll
      for (int i = 0; i < 4; ++i) d[tid + i * TPB] = s[tid + i * TPB];
    }
    __syncthreads();
#endif
    const u16* bufp = &pbuf[c & 1][0];
    for (int ptl = 0; ptl < CHUNK / 16; ++ptl) {
      int p0 = c * CHUNK + ptl * 16;
      const u16* bb = bufp + (size_t)ptl * 16 * 64;
      v8f acc = {};
      acc = wmma_bf16(a0, load_b16(bb,      64, lr, hf), acc);
      acc = wmma_bf16(a1, load_b16(bb + 32, 64, lr, hf), acc);
      union { uint4 q; u16 s[8]; } pk;
#pragma unroll
      for (int r = 0; r < 8; ++r) {
        float d = drow[(size_t)(r + hf * 8) * NP + p0 + lr];
        float v = acc[r] * 0.125f * band_mask(d, h);
        u16  ab = f2bf(__expf(v - mrow[r]) * invs[r]);
        att[(arow + r + hf * 8) * NP + p0 + lr] = ab;
        pk.s[r] = ab;
      }
      size_t toff = (((size_t)bh * NP) + p0 + lr) * NL + lt * 16 + hf * 8;
      *(uint4*)(att_t + toff) = pk.q;
    }
    __syncthreads();
  }
}

// ---------------------------------------------------------------------------
// l3 = att @ p2 : one wave per (b,h,ltile), 4 accumulators (full d=64),
// A (att row panel) loaded once per 4 WMMAs. K=2048 -> 256 WMMA/wave.
// ---------------------------------------------------------------------------
__global__ __launch_bounds__(TPB)
void l3_kernel(const u16* __restrict__ att, const u16* __restrict__ p2t,
               u16* __restrict__ l3cat) {
  int wid  = (blockIdx.x * TPB + threadIdx.x) >> 5;
  int lane = threadIdx.x & 31, lr = lane & 15, hf = lane >> 4;
  if (wid >= B_ * HH * 16) return;
  int lt = wid & 15, bh = wid >> 4;

  const u16* abase = att + (((size_t)bh * NL) + lt * 16) * NP;
  const u16* bb    = p2t + (size_t)bh * 64 * NP;
  v8f acc[4] = {};
  for (int ks = 0; ks < 64; ++ks) {
    if ((ks & 7) == 0 && ks + 8 < 64)
      __builtin_prefetch(abase + (size_t)lr * NP + (ks + 8) * 32, 0, 0);
    v16bf a = load_a16(abase + ks * 32, NP, lr, hf);
#pragma unroll
    for (int dt = 0; dt < 4; ++dt)
      acc[dt] = wmma_bf16(a, load_b16(bb + (size_t)dt * 16 * NP + ks * 32, NP, lr, hf),
                          acc[dt]);
  }
  int b = bh >> 3, h = bh & 7;
#pragma unroll
  for (int dt = 0; dt < 4; ++dt) {
    int n0 = h * 64 + dt * 16;
#pragma unroll
    for (int r = 0; r < 8; ++r)
      l3cat[((size_t)b * NL + lt * 16 + r + hf * 8) * 512 + n0 + lr] = f2bf(acc[dt][r]);
  }
}

// p3 = att^T @ l2 : one wave per (b,h,ptile), 4 accumulators, K=256.
__global__ __launch_bounds__(TPB)
void p3_kernel(const u16* __restrict__ att_t, const u16* __restrict__ l2t,
               u16* __restrict__ p3cat) {
  int wid  = (blockIdx.x * TPB + threadIdx.x) >> 5;
  int lane = threadIdx.x & 31, lr = lane & 15, hf = lane >> 4;
  if (wid >= B_ * HH * 128) return;
  int pt = wid & 127, bh = wid >> 7;

  const u16* abase = att_t + (((size_t)bh * NP) + pt * 16) * NL;
  const u16* bb    = l2t + (size_t)bh * 64 * NL;
  v8f acc[4] = {};
  for (int ks = 0; ks < 8; ++ks) {
    v16bf a = load_a16(abase + ks * 32, NL, lr, hf);
#pragma unroll
    for (int dt = 0; dt < 4; ++dt)
      acc[dt] = wmma_bf16(a, load_b16(bb + (size_t)dt * 16 * NL + ks * 32, NL, lr, hf),
                          acc[dt]);
  }
  int b = bh >> 3, h = bh & 7;
#pragma unroll
  for (int dt = 0; dt < 4; ++dt) {
    int n0 = h * 64 + dt * 16;
#pragma unroll
    for (int r = 0; r < 8; ++r)
      p3cat[((size_t)b * NP + pt * 16 + r + hf * 8) * 512 + n0 + lr] = f2bf(acc[dt][r]);
  }
}

// fc11/fc21: [M,512]@[512,64]+b (no relu) -> first half of concat buffer.
__global__ __launch_bounds__(TPB)
void fc_cat_kernel(const u16* __restrict__ A, const u16* __restrict__ wt,
                   const float* __restrict__ bias, u16* __restrict__ cat, int Mrows) {
  int wid  = (blockIdx.x * TPB + threadIdx.x) >> 5;
  int lane = threadIdx.x & 31, lr = lane & 15, hf = lane >> 4;
  if (wid >= Mrows / 16) return;
  const u16* abase = A + (size_t)wid * 16 * 512;
  v8f acc[4] = {};
  for (int ks = 0; ks < 16; ++ks) {
    v16bf a = load_a16(abase + ks * 32, 512, lr, hf);
#pragma unroll
    for (int nt = 0; nt < 4; ++nt)
      acc[nt] = wmma_bf16(a, load_b16(wt + (size_t)nt * 16 * 512 + ks * 32, 512, lr, hf),
                          acc[nt]);
  }
#pragma unroll
  for (int nt = 0; nt < 4; ++nt) {
    float bv = bias[nt * 16 + lr];
#pragma unroll
    for (int r = 0; r < 8; ++r)
      cat[(size_t)(wid * 16 + r + hf * 8) * 128 + nt * 16 + lr] = f2bf(acc[nt][r] + bv);
  }
}

// fc12/fc22: relu([M,128]@[128,64]+b) -> f32 d_out [B][2304][64].
__global__ __launch_bounds__(TPB)
void fc_out_kernel(const u16* __restrict__ cat, const u16* __restrict__ wt,
                   const float* __restrict__ bias, float* __restrict__ out,
                   int Mrows, int tok_pb, int row_off) {
  int wid  = (blockIdx.x * TPB + threadIdx.x) >> 5;
  int lane = threadIdx.x & 31, lr = lane & 15, hf = lane >> 4;
  if (wid >= Mrows / 16) return;
  const u16* abase = cat + (size_t)wid * 16 * 128;
  v8f acc[4] = {};
  for (int ks = 0; ks < 4; ++ks) {
    v16bf a = load_a16(abase + ks * 32, 128, lr, hf);
#pragma unroll
    for (int nt = 0; nt < 4; ++nt)
      acc[nt] = wmma_bf16(a, load_b16(wt + (size_t)nt * 16 * 128 + ks * 32, 128, lr, hf),
                          acc[nt]);
  }
#pragma unroll
  for (int nt = 0; nt < 4; ++nt) {
    float bv = bias[nt * 16 + lr];
#pragma unroll
    for (int r = 0; r < 8; ++r) {
      int row = wid * 16 + r + hf * 8;
      int b = row / tok_pb, t = row - b * tok_pb;
      out[((size_t)b * (NL + NP) + row_off + t) * 64 + nt * 16 + lr] =
          fmaxf(acc[nt][r] + bv, 0.0f);
    }
  }
}

// ===========================================================================
extern "C" void kernel_launch(void* const* d_in, const int* in_sizes, int n_in,
                              void* d_out, int out_size, void* d_ws, size_t ws_size,
                              hipStream_t stream) {
  const float* ligand = (const float*)d_in[0];
  const float* prot   = (const float*)d_in[1];
  const float* dist   = (const float*)d_in[2];
  const float* w_l1 = (const float*)d_in[3];  const float* b_l1 = (const float*)d_in[4];
  const float* w_l2 = (const float*)d_in[5];  const float* b_l2 = (const float*)d_in[6];
  const float* w_p1 = (const float*)d_in[7];  const float* b_p1 = (const float*)d_in[8];
  const float* w_p2 = (const float*)d_in[9];  const float* b_p2 = (const float*)d_in[10];
  const float* fc11_w = (const float*)d_in[11]; const float* fc11_b = (const float*)d_in[12];
  const float* fc12_w = (const float*)d_in[13]; const float* fc12_b = (const float*)d_in[14];
  const float* fc21_w = (const float*)d_in[15]; const float* fc21_b = (const float*)d_in[16];
  const float* fc22_w = (const float*)d_in[17]; const float* fc22_b = (const float*)d_in[18];
  float* out = (float*)d_out;

  char* p = (char*)d_ws;                       // ~392 MB total
  auto carve = [&](size_t bytes) -> char* {
    char* r = p; p += (bytes + 255) & ~(size_t)255; return r;
  };
  u16* cat1  = (u16*)carve((size_t)B_ * NL * 128 * 2);
  u16* cat2  = (u16*)carve((size_t)B_ * NP * 128 * 2);
  u16* wl1t  = (u16*)carve(512 * 64 * 2);
  u16* wl2t  = (u16*)carve(512 * 64 * 2);
  u16* wp1t  = (u16*)carve(512 * 64 * 2);
  u16* wp2t  = (u16*)carve(512 * 64 * 2);
  u16* f11t  = (u16*)carve(64 * 512 * 2);
  u16* f21t  = (u16*)carve(64 * 512 * 2);
  u16* f12t  = (u16*)carve(64 * 128 * 2);
  u16* f22t  = (u16*)carve(64 * 128 * 2);
  u16* l1h   = (u16*)carve((size_t)B_ * HH * NL * 64 * 2);
  u16* p1h   = (u16*)carve((size_t)B_ * HH * NP * 64 * 2);
  u16* l2t   = (u16*)carve((size_t)B_ * HH * 64 * NL * 2);
  u16* p2t   = (u16*)carve((size_t)B_ * HH * 64 * NP * 2);
  float* stats = (float*)carve((size_t)B_ * HH * NL * 2 * 4);
  u16* att   = (u16*)carve((size_t)B_ * HH * NL * NP * 2);   // 134 MB, L2-resident
  u16* att_t = (u16*)carve((size_t)B_ * HH * NP * NL * 2);
  u16* l3cat = (u16*)carve((size_t)B_ * NL * 512 * 2);
  u16* p3cat = (u16*)carve((size_t)B_ * NP * 512 * 2);

  dim3 blk(TPB);
  auto gwav = [](int waves) { return dim3((unsigned)((waves + 7) / 8)); };

  cvt_cat_kernel<<<dim3((B_ * NL * 64 + TPB - 1) / TPB), blk, 0, stream>>>(ligand, cat1, B_ * NL * 64);
  cvt_cat_kernel<<<dim3((B_ * NP * 64 + TPB - 1) / TPB), blk, 0, stream>>>(prot,   cat2, B_ * NP * 64);

  wt_t_kernel<<<dim3((64 * 512 + TPB - 1) / TPB), blk, 0, stream>>>(w_l1, wl1t, 64, 512);
  wt_t_kernel<<<dim3((64 * 512 + TPB - 1) / TPB), blk, 0, stream>>>(w_l2, wl2t, 64, 512);
  wt_t_kernel<<<dim3((64 * 512 + TPB - 1) / TPB), blk, 0, stream>>>(w_p1, wp1t, 64, 512);
  wt_t_kernel<<<dim3((64 * 512 + TPB - 1) / TPB), blk, 0, stream>>>(w_p2, wp2t, 64, 512);
  wt_t_kernel<<<dim3((512 * 64 + TPB - 1) / TPB), blk, 0, stream>>>(fc11_w, f11t, 512, 64);
  wt_t_kernel<<<dim3((512 * 64 + TPB - 1) / TPB), blk, 0, stream>>>(fc21_w, f21t, 512, 64);
  wt_t_kernel<<<dim3((128 * 64 + TPB - 1) / TPB), blk, 0, stream>>>(fc12_w, f12t, 128, 64);
  wt_t_kernel<<<dim3((128 * 64 + TPB - 1) / TPB), blk, 0, stream>>>(fc22_w, f22t, 128, 64);

  proj_kernel<<<gwav(B_ * (NL / 16) * HH), blk, 0, stream>>>(cat1 + 64, wl1t, b_l1, NL, l1h, nullptr);
  proj_kernel<<<gwav(B_ * (NL / 16) * HH), blk, 0, stream>>>(cat1 + 64, wl2t, b_l2, NL, nullptr, l2t);
  proj_kernel<<<gwav(B_ * (NP / 16) * HH), blk, 0, stream>>>(cat2 + 64, wp1t, b_p1, NP, p1h, nullptr);
  proj_kernel<<<gwav(B_ * (NP / 16) * HH), blk, 0, stream>>>(cat2 + 64, wp2t, b_p2, NP, nullptr, p2t);

  att_stats_kernel<<<dim3(B_ * HH * 2), blk, 0, stream>>>(l1h, p1h, dist, stats);
  att_write_kernel<<<dim3(B_ * HH * 2), blk, 0, stream>>>(l1h, p1h, dist, stats, att, att_t);

  l3_kernel<<<gwav(B_ * HH * 16), blk, 0, stream>>>(att, p2t, l3cat);
  p3_kernel<<<gwav(B_ * HH * 128), blk, 0, stream>>>(att_t, l2t, p3cat);

  fc_cat_kernel<<<gwav(B_ * NL / 16), blk, 0, stream>>>(l3cat, f11t, fc11_b, cat1, B_ * NL);
  fc_cat_kernel<<<gwav(B_ * NP / 16), blk, 0, stream>>>(p3cat, f21t, fc21_b, cat2, B_ * NP);
  fc_out_kernel<<<gwav(B_ * NL / 16), blk, 0, stream>>>(cat1, f12t, fc12_b, out, B_ * NL, NL, 0);
  fc_out_kernel<<<gwav(B_ * NP / 16), blk, 0, stream>>>(cat2, f22t, fc22_b, out, B_ * NP, NP, NL);
}
